// DVSFFNet_62723702391146
// MI455X (gfx1250) — matrix operand order repaired and checked by reading, not compiled, not run on hardware
//
#include <hip/hip_runtime.h>

typedef _Float16 half16 __attribute__((ext_vector_type(16)));
typedef _Float16 half8  __attribute__((ext_vector_type(8)));
typedef float    float8 __attribute__((ext_vector_type(8)));

#define CCH 128      // channels
#define TT  16       // time steps
#define NN  4        // batch
#define TN  64       // T*N
#define KCONV 1152   // 128*9 im2col depth
#define LDSROW 1160  // padded LDS row stride in halves (580 dwords = 4 mod 64 banks)
#define LDSBYTES (CCH * LDSROW * 2)   // 296960 B <= 320KB WGP LDS

// ---------------- BN prep: inv = gm/sqrt(vr+eps), add = bt - mu*inv ----------------
__global__ void bnprep_kernel(const float* __restrict__ gm, const float* __restrict__ bt,
                              const float* __restrict__ mu, const float* __restrict__ vr,
                              float* __restrict__ inv, float* __restrict__ add) {
  int c = threadIdx.x;
  if (c < CCH) {
    float iv = gm[c] * rsqrtf(vr[c] + 1e-5f);
    inv[c] = iv;
    add[c] = bt[c] - mu[c] * iv;
  }
}

// ------------- Pack conv weight [128,128,3,3] OIHW -> [co][k] fp16, k=(kh*3+kw)*128+ci -------------
__global__ void pack_convw_kernel(const float* __restrict__ w, _Float16* __restrict__ out) {
  int idx = blockIdx.x * blockDim.x + threadIdx.x;
  if (idx >= CCH * KCONV) return;
  int m = idx / KCONV, k = idx - m * KCONV;
  int tap = k >> 7, ci = k & 127;
  int kh = (tap * 11) >> 5, kw = tap - kh * 3;
  out[idx] = (_Float16)w[m * KCONV + ci * 9 + kh * 3 + kw];
}

// ------------- Pack fc1 [512, 2048(f=c*16+hw)] -> [o][k'=hw*128+c] fp16 -------------
__global__ void pack_fc1_kernel(const float* __restrict__ w, _Float16* __restrict__ out) {
  int idx = blockIdx.x * blockDim.x + threadIdx.x;
  if (idx >= 512 * 2048) return;
  int m = idx >> 11, k = idx & 2047;
  int hw = k >> 7, c = k & 127;
  out[idx] = (_Float16)w[(m << 11) + c * 16 + hw];
}

// ------------- Pack fc2 [110,512] -> [112,512] fp16 (zero-padded rows) + padded bias -------------
__global__ void pack_fc2_kernel(const float* __restrict__ w, const float* __restrict__ b,
                                _Float16* __restrict__ out, float* __restrict__ bp) {
  int idx = blockIdx.x * blockDim.x + threadIdx.x;
  if (idx >= 112 * 512) return;
  int m = idx >> 9, k = idx & 511;
  out[idx] = (m < 110) ? (_Float16)w[m * 512 + k] : (_Float16)0.f;
  if (k == 0) bp[m] = (m < 110) ? b[m] : 0.f;
}

// ------------- Layer 0 fused: conv3x3(Cin=2)+BN+LIF(T)+pool -> spikes [64,64,64,128] fp16 -------------
__global__ void conv0_lif_pool_kernel(const float* __restrict__ X,   // [4,16,2,128,128]
                                      const float* __restrict__ W0,  // [128,2,3,3]
                                      const float* __restrict__ inv,
                                      const float* __restrict__ add,
                                      _Float16* __restrict__ Sout) { // [64,64,64,128] ch-last
  int idx = blockIdx.x * blockDim.x + threadIdx.x;
  if (idx >= NN * 64 * 64 * CCH) return;
  int co = idx & 127;
  int r  = idx >> 7;
  int w2 = r & 63; r >>= 6;
  int h2 = r & 63;
  int n  = r >> 6;
  float wreg[18];
#pragma unroll
  for (int i = 0; i < 18; ++i) wreg[i] = W0[co * 18 + i];
  float cinv = inv[co], cadd = add[co];
  float v[4] = {0.f, 0.f, 0.f, 0.f};
  for (int t = 0; t < TT; ++t) {
    const float* xp = X + (((long)n * TT + t) * 2) * 16384;
    float smax = 0.f;
#pragma unroll
    for (int q = 0; q < 4; ++q) {
      int ph = 2 * h2 + (q >> 1), pw = 2 * w2 + (q & 1);
      float acc = 0.f;
#pragma unroll
      for (int c = 0; c < 2; ++c)
#pragma unroll
        for (int kh = 0; kh < 3; ++kh) {
          int sh = ph + kh - 1;
          if ((unsigned)sh < 128u) {
#pragma unroll
            for (int kw = 0; kw < 3; ++kw) {
              int sw = pw + kw - 1;
              if ((unsigned)sw < 128u)
                acc += xp[c * 16384 + sh * 128 + sw] * wreg[c * 9 + kh * 3 + kw];
            }
          }
        }
      float y = acc * cinv + cadd;
      v[q] = 0.5f * (v[q] + y);                 // v += (x - v)/tau, tau=2, reset=0
      float s = (v[q] >= 1.f) ? 1.f : 0.f;      // atan_spike fwd: v - 1 >= 0
      v[q] *= (1.f - s);                        // hard reset
      smax = fmaxf(smax, s);
    }
    int tn = t * NN + n;
    Sout[((long)tn * 4096 + h2 * 64 + w2) * CCH + co] = (_Float16)smax;
  }
}

// ------------- WMMA conv (layers 1-4): S[TN,H,H,128] fp16 -> Y[TN,H,H,128] fp16 (BN fused) -------------
// Block = 512 threads (16 waves). The whole 288KB packed weight matrix is staged once into
// LDS (padded rows -> conflict-free ds_load_b128) and shared by all 16 waves. Each wave
// computes ALL 128 output channels of one 16-pixel tile: one 32B global B-gather per K-step
// feeds 8 back-to-back v_wmma_f32_16x16x32_f16.
template <int H>
__global__ __launch_bounds__(512) void conv_wmma_kernel(
    const _Float16* __restrict__ S,
    const _Float16* __restrict__ Wpk,   // [128,1152]
    const float* __restrict__ inv,
    const float* __restrict__ add,
    _Float16* __restrict__ Y) {
  constexpr int W  = H;
  constexpr int HW = H * W;
  constexpr int PT = HW >> 4;   // pixel tiles (power of two)
  extern __shared__ _Float16 Wlds[];   // [128][LDSROW]

  // --- cooperative stage of the full weight panel: 18432 x 16B chunks ---
  constexpr int CHUNKS = CCH * (KCONV / 8);   // 18432
  for (int c = threadIdx.x; c < CHUNKS; c += 512) {
    int m  = c / (KCONV / 8);          // 144 chunks per row
    int kc = c - m * (KCONV / 8);
    *(half8*)(Wlds + m * LDSROW + kc * 8) = *(const half8*)(Wpk + m * KCONV + kc * 8);
  }
  __syncthreads();

  int wave = blockIdx.x * 16 + (threadIdx.x >> 5);
  int lane = threadIdx.x & 31;
  int pt = wave & (PT - 1);
  int tn = wave / PT;           // PT is power of two -> shift
  int col = lane & 15;
  int hi  = lane >> 4;
  int p  = pt * 16 + col;
  int ph = p / W, pw = p & (W - 1);   // W power of two
  const _Float16* Srow = S + (size_t)tn * HW * CCH;
  const _Float16* Albase = Wlds + (size_t)col * LDSROW;
  float8 acc[8] = {};
#pragma unroll 4
  for (int kb = 0; kb < KCONV; kb += 32) {
    // --- B fragment: lane covers k0..k0+15, always inside ONE tap -> contiguous 32B ---
    int k0  = kb + hi * 16;
    int tap = k0 >> 7;
    int ci0 = k0 & 127;
    int kh  = (tap * 11) >> 5;          // tap/3 for tap in [0,9)
    int kw  = tap - kh * 3;
    int sh  = ph + kh - 1, sw = pw + kw - 1;
    half16 b = {};
    if ((unsigned)sh < (unsigned)H && (unsigned)sw < (unsigned)W)
      b = *(const half16*)(Srow + (sh * W + sw) * CCH + ci0);
    // --- 8 channel tiles: A fragment from LDS (2x ds_load_b128), back-to-back WMMAs ---
#pragma unroll
    for (int ct = 0; ct < 8; ++ct) {
      const _Float16* wr = Albase + (size_t)(ct * 16) * LDSROW + kb + hi * 16;
      half8 a0 = *(const half8*)wr;
      half8 a1 = *(const half8*)(wr + 8);
      half16 a = __builtin_shufflevector(a0, a1, 0, 1, 2, 3, 4, 5, 6, 7,
                                         8, 9, 10, 11, 12, 13, 14, 15);
      acc[ct] = __builtin_amdgcn_wmma_f32_16x16x32_f16(false, a, false, b, (short)0,
                                                       acc[ct], false, false);
    }
  }
  // Epilogue: BN + f32->f16; each lane stores 8x 16B of contiguous channels
  _Float16* yo = Y + ((size_t)tn * HW + p) * CCH;
#pragma unroll
  for (int ct = 0; ct < 8; ++ct) {
    int mbase = ct * 16 + hi * 8;
    half8 o;
#pragma unroll
    for (int rr = 0; rr < 8; ++rr)
      o[rr] = (_Float16)(acc[ct][rr] * inv[mbase + rr] + add[mbase + rr]);
    *(half8*)(yo + mbase) = o;
  }
}

// ------------- LIF(T) + 2x2 pool: Y[TN,H,H,128] -> S[TN,H/2,H/2,128] fp16 -------------
template <int H>
__global__ void lif_pool_kernel(const _Float16* __restrict__ Yin,
                                _Float16* __restrict__ Sout) {
  constexpr int W = H, H2 = H >> 1, W2 = W >> 1;
  constexpr int total = NN * H2 * W2 * CCH;
  int idx = blockIdx.x * blockDim.x + threadIdx.x;
  if (idx >= total) return;
  int co = idx & 127;
  int r  = idx >> 7;
  int w2 = r & (W2 - 1); r /= W2;
  int h2 = r & (H2 - 1);
  int n  = r / H2;
  long o00 = ((long)(2 * h2) * W + 2 * w2) * CCH + co;
  long o01 = o00 + CCH;
  long o10 = o00 + (long)W * CCH;
  long o11 = o10 + CCH;
  constexpr long imgS = (long)H * W * CCH;
  constexpr long outS = (long)H2 * W2 * CCH;
  long oo = ((long)h2 * W2 + w2) * CCH + co;
  float v0 = 0.f, v1 = 0.f, v2 = 0.f, v3 = 0.f;
  for (int t = 0; t < TT; ++t) {
    int tn = t * NN + n;
    const _Float16* yp = Yin + tn * imgS;
    float s, smax = 0.f;
    v0 = 0.5f * (v0 + (float)yp[o00]); s = (v0 >= 1.f) ? 1.f : 0.f; v0 *= (1.f - s); smax = fmaxf(smax, s);
    v1 = 0.5f * (v1 + (float)yp[o01]); s = (v1 >= 1.f) ? 1.f : 0.f; v1 *= (1.f - s); smax = fmaxf(smax, s);
    v2 = 0.5f * (v2 + (float)yp[o10]); s = (v2 >= 1.f) ? 1.f : 0.f; v2 *= (1.f - s); smax = fmaxf(smax, s);
    v3 = 0.5f * (v3 + (float)yp[o11]); s = (v3 >= 1.f) ? 1.f : 0.f; v3 *= (1.f - s); smax = fmaxf(smax, s);
    Sout[tn * outS + oo] = (_Float16)smax;
  }
}

// ------------- WMMA FC GEMM: Y[64,M] = Sin[64,K] * Wpk[M,K]^T + bias -------------
template <int M, int K>
__global__ void fc_wmma_kernel(const _Float16* __restrict__ Sin,  // [64,K] row-major
                               const _Float16* __restrict__ Wpk,  // [M,K]
                               const float* __restrict__ bias,    // [M]
                               float* __restrict__ Yout) {        // [64,M]
  constexpr int MT = M >> 4;
  int wave = blockIdx.x * (blockDim.x >> 5) + (threadIdx.x >> 5);
  int lane = threadIdx.x & 31;
  int nt = wave / MT;     // 0..3 (tn tiles)
  int mt = wave % MT;
  int col = lane & 15, hi = lane >> 4;
  int tn = nt * 16 + col;
  const _Float16* Arow = Wpk + (size_t)(mt * 16 + col) * K;
  const _Float16* Brow = Sin + (size_t)tn * K;
  float8 acc = {};
#pragma unroll 4
  for (int kb = 0; kb < K; kb += 32) {
    const _Float16* wr = Arow + kb + hi * 16;
    half8 a0 = *(const half8*)wr;
    half8 a1 = *(const half8*)(wr + 8);
    half16 a = __builtin_shufflevector(a0, a1, 0, 1, 2, 3, 4, 5, 6, 7,
                                       8, 9, 10, 11, 12, 13, 14, 15);
    half16 b = *(const half16*)(Brow + kb + hi * 16);
    acc = __builtin_amdgcn_wmma_f32_16x16x32_f16(false, a, false, b, (short)0, acc,
                                                 false, false);
  }
  int mbase = mt * 16 + hi * 8;
  float* yo = Yout + (size_t)tn * M + mbase;
#pragma unroll
  for (int rr = 0; rr < 8; ++rr) yo[rr] = acc[rr] + bias[mbase + rr];
}

// ------------- LIF over T for FC outputs: Yin[64,M] f32 -> spikes [64,M] fp16 -------------
__global__ void lif_fc_kernel(const float* __restrict__ Yin, _Float16* __restrict__ Sout, int M) {
  int idx = blockIdx.x * blockDim.x + threadIdx.x;
  if (idx >= NN * M) return;
  int o = idx % M, n = idx / M;
  float v = 0.f;
  for (int t = 0; t < TT; ++t) {
    int tn = t * NN + n;
    v = 0.5f * (v + Yin[(long)tn * M + o]);
    float s = (v >= 1.f) ? 1.f : 0.f;
    v *= (1.f - s);
    Sout[(long)tn * M + o] = (_Float16)s;
  }
}

// ------------- Final: LIF over fc2 pre-acts + VotingLayer(10) + rate readout -> out[4,11] -------------
__global__ void vote_kernel(const float* __restrict__ Y2,  // [64,112]
                            float* __restrict__ out) {     // [4,11]
  int idx = threadIdx.x;
  if (idx >= NN * 11) return;
  int j = idx % 11, n = idx / 11;
  float acc = 0.f;
  for (int i = 0; i < 10; ++i) {
    float v = 0.f;
    for (int t = 0; t < TT; ++t) {
      float x = Y2[(long)(t * NN + n) * 112 + j * 10 + i];
      v = 0.5f * (v + x);
      float s = (v >= 1.f) ? 1.f : 0.f;
      v *= (1.f - s);
      acc += s;
    }
  }
  out[idx] = acc * (1.f / 160.f);   // mean over 10 votes and 16 time steps
}

// =======================================================================================
extern "C" void kernel_launch(void* const* d_in, const int* in_sizes, int n_in,
                              void* d_out, int out_size, void* d_ws, size_t ws_size,
                              hipStream_t stream) {
  const float* x = (const float*)d_in[0];
  const float* w[5];
  const float *gm[5], *bt[5], *mu[5], *vr[5];
  for (int l = 0; l < 5; ++l) {
    w[l]  = (const float*)d_in[1 + 5 * l];
    gm[l] = (const float*)d_in[2 + 5 * l];
    bt[l] = (const float*)d_in[3 + 5 * l];
    mu[l] = (const float*)d_in[4 + 5 * l];
    vr[l] = (const float*)d_in[5 + 5 * l];
  }
  const float* fc1_w = (const float*)d_in[26];
  const float* fc1_b = (const float*)d_in[27];
  const float* fc2_w = (const float*)d_in[28];
  const float* fc2_b = (const float*)d_in[29];

  char* ws = (char*)d_ws;
  size_t off = 0;
  auto alloc = [&](size_t bytes) -> void* {
    off = (off + 255) & ~(size_t)255;
    void* p = ws + off;
    off += bytes;
    return p;
  };

  float *invb[5], *addb[5];
  for (int l = 0; l < 5; ++l) {
    invb[l] = (float*)alloc(CCH * 4);
    addb[l] = (float*)alloc(CCH * 4);
  }
  _Float16* wpk[4];
  for (int l = 0; l < 4; ++l) wpk[l] = (_Float16*)alloc((size_t)CCH * KCONV * 2);
  _Float16* fc1p = (_Float16*)alloc((size_t)512 * 2048 * 2);
  _Float16* fc2p = (_Float16*)alloc((size_t)112 * 512 * 2);
  float*    b2p  = (float*)alloc(112 * 4);

  _Float16* s0 = (_Float16*)alloc((size_t)TN * 64 * 64 * CCH * 2);  // 67 MB
  _Float16* yb = (_Float16*)alloc((size_t)TN * 64 * 64 * CCH * 2);  // 67 MB ping
  _Float16* s1 = (_Float16*)alloc((size_t)TN * 32 * 32 * CCH * 2);
  _Float16* s2 = (_Float16*)alloc((size_t)TN * 16 * 16 * CCH * 2);
  _Float16* s3 = (_Float16*)alloc((size_t)TN * 8 * 8 * CCH * 2);
  _Float16* s4 = (_Float16*)alloc((size_t)TN * 4 * 4 * CCH * 2);
  float*    yfc1 = (float*)alloc((size_t)TN * 512 * 4);
  _Float16* sfc1 = (_Float16*)alloc((size_t)TN * 512 * 2);
  float*    yfc2 = (float*)alloc((size_t)TN * 112 * 4);

  // --- parameter prep ---
  for (int l = 0; l < 5; ++l)
    bnprep_kernel<<<1, 128, 0, stream>>>(gm[l], bt[l], mu[l], vr[l], invb[l], addb[l]);
  for (int l = 1; l < 5; ++l)
    pack_convw_kernel<<<(CCH * KCONV + 255) / 256, 256, 0, stream>>>(w[l], wpk[l - 1]);
  pack_fc1_kernel<<<(512 * 2048 + 255) / 256, 256, 0, stream>>>(fc1_w, fc1p);
  pack_fc2_kernel<<<(112 * 512 + 255) / 256, 256, 0, stream>>>(fc2_w, fc2_b, fc2p, b2p);

  // --- layer 0 (fused, VALU) ---
  conv0_lif_pool_kernel<<<(NN * 64 * 64 * CCH + 255) / 256, 256, 0, stream>>>(
      x, w[0], invb[0], addb[0], s0);

  // --- layers 1-4: WMMA conv (16 waves/block, full weight panel in LDS) + LIF/pool ---
  conv_wmma_kernel<64><<<TN * 256 / 16, 512, LDSBYTES, stream>>>(s0, wpk[0], invb[1], addb[1], yb);
  lif_pool_kernel<64><<<(NN * 32 * 32 * CCH + 255) / 256, 256, 0, stream>>>(yb, s1);

  conv_wmma_kernel<32><<<TN * 64 / 16, 512, LDSBYTES, stream>>>(s1, wpk[1], invb[2], addb[2], yb);
  lif_pool_kernel<32><<<(NN * 16 * 16 * CCH + 255) / 256, 256, 0, stream>>>(yb, s2);

  conv_wmma_kernel<16><<<TN * 16 / 16, 512, LDSBYTES, stream>>>(s2, wpk[2], invb[3], addb[3], yb);
  lif_pool_kernel<16><<<(NN * 8 * 8 * CCH + 255) / 256, 256, 0, stream>>>(yb, s3);

  conv_wmma_kernel<8><<<TN * 4 / 16, 512, LDSBYTES, stream>>>(s3, wpk[3], invb[4], addb[4], yb);
  lif_pool_kernel<8><<<(NN * 4 * 4 * CCH + 255) / 256, 256, 0, stream>>>(yb, s4);

  // --- FC1 (M=512, K=2048) + LIF ---
  fc_wmma_kernel<512, 2048><<<(4 * 32) / 4, 128, 0, stream>>>(s4, fc1p, fc1_b, yfc1);
  lif_fc_kernel<<<(NN * 512 + 255) / 256, 256, 0, stream>>>(yfc1, sfc1, 512);

  // --- FC2 (M=112 padded, K=512) + LIF + voting ---
  fc_wmma_kernel<112, 512><<<(4 * 7) / 4, 128, 0, stream>>>(sfc1, fc2p, b2p, yfc2);
  vote_kernel<<<1, 64, 0, stream>>>(yfc2, (float*)d_out);
}
